// GCN_cls_e_69483980915280
// MI455X (gfx1250) — matrix operand order repaired and checked by reading, not compiled, not run on hardware
//
#include <hip/hip_runtime.h>

typedef __attribute__((ext_vector_type(16))) _Float16 v16h;
typedef __attribute__((ext_vector_type(8)))  _Float16 v8h;
typedef __attribute__((ext_vector_type(8)))  float    v8f;

#define N_NODES 100000
#define N_EDGES 1600000
#define F_INF   50
#define HID     64
#define N_GRAPH 512
#define N_CLS   10
#define BOND    10

// ---------- order-preserving float<->uint for atomic max ----------
__device__ __forceinline__ unsigned f2ord(float f) {
  unsigned u = __float_as_uint(f);
  return (u & 0x80000000u) ? ~u : (u | 0x80000000u);
}
__device__ __forceinline__ float ord2f(unsigned u) {
  return (u & 0x80000000u) ? __uint_as_float(u & 0x7FFFFFFFu) : __uint_as_float(~u);
}

// ---------- WMMA tile loaders: branch-free, two b128 loads per operand ----------
// A 16x32 f16 (ISA 7.12.2): lane<16 row m=lane, K={kg..kg+7, kg+16..kg+23}, kg = (lane>>4)*8.
// Element i<8 -> K=kg+i ; element i>=8 -> K=kg+16+(i-8): two contiguous 8-half runs.
__device__ __forceinline__ v16h load_a_tile(const _Float16* __restrict__ A, int row0,
                                            int lda, int k0) {
  const int lane = threadIdx.x & 31;
  const int kg   = (lane >> 4) * 8;
  const int m    = lane & 15;
  const _Float16* p = A + (size_t)(row0 + m) * lda + k0 + kg;
  v8h lo = *(const v8h*)(p);
  v8h hi = *(const v8h*)(p + 16);
  v16h a;
#pragma unroll
  for (int i = 0; i < 8; ++i) { a[i] = lo[i]; a[i + 8] = hi[i]; }
  return a;
}

// B 32x16 f16 = W^T tile, W row-major f16 [ncols, ldk]: lane holds col n=lane&15,
// K = k0 + kg + i (i<8), k0 + kg + 16 + (i-8) (i>=8), kg=(lane>>4)*8: two b128 loads.
__device__ __forceinline__ v16h load_b_tile(const _Float16* __restrict__ W, int col0,
                                            int ldk, int k0) {
  const int lane = threadIdx.x & 31;
  const int kg   = (lane >> 4) * 8;
  const int n    = lane & 15;
  const _Float16* p = W + (size_t)(col0 + n) * ldk + k0 + kg;
  v8h lo = *(const v8h*)(p);
  v8h hi = *(const v8h*)(p + 16);
  v16h b;
#pragma unroll
  for (int i = 0; i < 8; ++i) { b[i] = lo[i]; b[i + 8] = hi[i]; }
  return b;
}

__device__ __forceinline__ v8f wmma_acc(v16h a, v16h b, v8f c) {
  return __builtin_amdgcn_wmma_f32_16x16x32_f16(false, a, false, b, (short)0, c, false, false);
}

// ---------- kernels ----------
__global__ void __launch_bounds__(256) zero_kernel(float* __restrict__ p, int n) {
  int i = blockIdx.x * blockDim.x + threadIdx.x;
  if (i < n) p[i] = 0.0f;
}

// f32 [srows, scols] -> f16 [total/dcols, dcols], zero-padded, optional relu
__global__ void __launch_bounds__(256) cvt_f16_kernel(const float* __restrict__ src,
                                                      _Float16* __restrict__ dst,
                                                      int srows, int scols, int dcols,
                                                      int relu, int total) {
  int t = blockIdx.x * blockDim.x + threadIdx.x;
  if (t >= total) return;
  int r = t / dcols, c = t - r * dcols;
  float v = (r < srows && c < scols) ? src[(size_t)r * scols + c] : 0.0f;
  if (relu) v = fmaxf(v, 0.0f);
  dst[t] = (_Float16)v;
}

__global__ void __launch_bounds__(256) edge_w_kernel(const float* __restrict__ ea,
                                                     const float* __restrict__ bw,
                                                     const float* __restrict__ bb,
                                                     float* __restrict__ ew) {
  int e = blockIdx.x * blockDim.x + threadIdx.x;
  if (e >= N_EDGES) return;
  const float* row = ea + (size_t)e * BOND;
  float acc = bb[0];
#pragma unroll
  for (int i = 0; i < BOND; ++i) acc += row[i] * bw[i];
  ew[e] = acc;
}

// thread per (edge, feature); lanes cover features -> coalesced row bursts, L2-resident atomics
__global__ void __launch_bounds__(256) scatter1_kernel(const float* __restrict__ x,
                                                       const int* __restrict__ ei,
                                                       const float* __restrict__ ew,
                                                       float* __restrict__ agg) {
  int t = blockIdx.x * blockDim.x + threadIdx.x;
  int e = t >> 6, f = t & 63;
  if (e >= N_EDGES || f >= F_INF) return;
  int s = ei[e];
  int d = ei[N_EDGES + e];
  atomicAdd(&agg[(size_t)d * F_INF + f], x[(size_t)s * F_INF + f] * ew[e]);
}

__global__ void __launch_bounds__(256) scatter2_kernel(const float* __restrict__ h,
                                                       const int* __restrict__ ei,
                                                       const float* __restrict__ ew,
                                                       float* __restrict__ agg) {
  int t = blockIdx.x * blockDim.x + threadIdx.x;
  int e = t >> 6, f = t & 63;
  if (e >= N_EDGES) return;
  int s = ei[e];
  int d = ei[N_EDGES + e];
  float v = fmaxf(h[(size_t)s * HID + f], 0.0f);  // relu fused into gather
  atomicAdd(&agg[(size_t)d * HID + f], v * ew[e]);
}

// out[N,64] = agg@rel_w.T + rel_b + xin@root_w.T ; all f16 inputs K-padded to 64
__global__ void __launch_bounds__(256) dense_conv_kernel(const _Float16* __restrict__ agg,
                                                         const _Float16* __restrict__ xin,
                                                         const _Float16* __restrict__ rel_w,
                                                         const float* __restrict__ rel_b,
                                                         const _Float16* __restrict__ root_w,
                                                         float* __restrict__ out) {
  const int gw   = blockIdx.x * 8 + (threadIdx.x >> 5);
  const int row0 = (gw >> 2) * 16;
  const int col0 = (gw & 3) * 16;
  const int lane = threadIdx.x & 31;
  v8f c = {};
#pragma unroll
  for (int k0 = 0; k0 < 64; k0 += 32) {
    c = wmma_acc(load_a_tile(agg, row0, 64, k0), load_b_tile(rel_w, col0, 64, k0), c);
    c = wmma_acc(load_a_tile(xin, row0, 64, k0), load_b_tile(root_w, col0, 64, k0), c);
  }
  const int kgrp = lane >> 4;
  const int n    = col0 + (lane & 15);
  const float bias = rel_b[n];
#pragma unroll
  for (int r = 0; r < 8; ++r)
    out[(size_t)(row0 + r + 8 * kgrp) * HID + n] = c[r] + bias;
}

// out = (relu?)(in@W.T + b); W padded so B loads are unguarded; f16 or f32 output
__global__ void __launch_bounds__(256) linear_wmma_kernel(const _Float16* __restrict__ in,
                                                          const _Float16* __restrict__ W,
                                                          const float* __restrict__ bias,
                                                          float* __restrict__ outf,
                                                          _Float16* __restrict__ outh,
                                                          int kdim, int ncols_store,
                                                          int ostride, int col_tiles,
                                                          int relu_out) {
  const int gw   = blockIdx.x * 8 + (threadIdx.x >> 5);
  const int row0 = (gw / col_tiles) * 16;
  const int col0 = (gw % col_tiles) * 16;
  const int lane = threadIdx.x & 31;
  v8f c = {};
  for (int k0 = 0; k0 < kdim; k0 += 32)
    c = wmma_acc(load_a_tile(in, row0, kdim, k0), load_b_tile(W, col0, kdim, k0), c);
  const int kgrp = lane >> 4;
  const int n    = col0 + (lane & 15);
  if (n < ncols_store) {
    const float b = bias[n];
#pragma unroll
    for (int r = 0; r < 8; ++r) {
      float v = c[r] + b;
      if (relu_out) v = fmaxf(v, 0.0f);
      size_t idx = (size_t)(row0 + r + 8 * kgrp) * ostride + n;
      if (outh) outh[idx] = (_Float16)v;
      else      outf[idx] = v;
    }
  }
}

__global__ void __launch_bounds__(256) pool_kernel(const float* __restrict__ h,
                                                   const int* __restrict__ batch,
                                                   unsigned* __restrict__ pmax,
                                                   float* __restrict__ psum,
                                                   float* __restrict__ cnt, int add_cnt) {
  int t = blockIdx.x * blockDim.x + threadIdx.x;  // N*64
  if (t >= N_NODES * HID) return;
  int nd = t >> 6, f = t & 63;
  int g = batch[nd];
  float v = h[t];
  atomicMax(&pmax[g * HID + f], f2ord(v));
  atomicAdd(&psum[g * HID + f], v);
  if (add_cnt && f == 0) atomicAdd(&cnt[g], 1.0f);
}

__global__ void __launch_bounds__(256) combine_kernel(const unsigned* __restrict__ pmax1,
                                                      const float* __restrict__ psum1,
                                                      const unsigned* __restrict__ pmax2,
                                                      const float* __restrict__ psum2,
                                                      const float* __restrict__ cnt,
                                                      _Float16* __restrict__ gsumh) {
  int t = blockIdx.x * blockDim.x + threadIdx.x;  // G*64
  if (t >= N_GRAPH * HID) return;
  int g = t >> 6, f = t & 63;
  float c = fmaxf(cnt[g], 1.0f);
  gsumh[(size_t)g * (2 * HID) + f]       = (_Float16)(ord2f(pmax1[t]) + ord2f(pmax2[t]));
  gsumh[(size_t)g * (2 * HID) + HID + f] = (_Float16)(psum1[t] / c + psum2[t] / c);
}

extern "C" void kernel_launch(void* const* d_in, const int* in_sizes, int n_in,
                              void* d_out, int out_size, void* d_ws, size_t ws_size,
                              hipStream_t stream) {
  (void)in_sizes; (void)n_in; (void)out_size; (void)ws_size;
  const float* x         = (const float*)d_in[0];
  const float* edge_a    = (const float*)d_in[1];
  const float* bond_w    = (const float*)d_in[2];
  const float* bond_b    = (const float*)d_in[3];
  const float* c1_rel_w  = (const float*)d_in[4];
  const float* c1_rel_b  = (const float*)d_in[5];
  const float* c1_root_w = (const float*)d_in[6];
  const float* c2_rel_w  = (const float*)d_in[7];
  const float* c2_rel_b  = (const float*)d_in[8];
  const float* c2_root_w = (const float*)d_in[9];
  const float* lin1_w    = (const float*)d_in[10];
  const float* lin1_b    = (const float*)d_in[11];
  const float* lin2_w    = (const float*)d_in[12];
  const float* lin2_b    = (const float*)d_in[13];
  const int*   eidx      = (const int*)d_in[14];
  const int*   batch     = (const int*)d_in[15];
  float* out = (float*)d_out;

  // ---------------- workspace layout ----------------
  float* ws = (float*)d_ws;
  size_t o = 0;
  float* ew = ws + o; o += (size_t)N_EDGES;
  float* h1 = ws + o; o += (size_t)N_NODES * HID;
  float* h2 = ws + o; o += (size_t)N_NODES * HID;
  // contiguous zeroed region
  float*    zbase = ws + o;
  float*    agg1  = zbase;
  float*    agg2  = agg1 + (size_t)N_NODES * F_INF;
  unsigned* pmax1 = (unsigned*)(agg2 + (size_t)N_NODES * HID);
  float*    psum1 = (float*)(pmax1 + (size_t)N_GRAPH * HID);
  unsigned* pmax2 = (unsigned*)(psum1 + (size_t)N_GRAPH * HID);
  float*    psum2 = (float*)(pmax2 + (size_t)N_GRAPH * HID);
  float*    cnt   = psum2 + (size_t)N_GRAPH * HID;
  const int nzero = (int)((size_t)N_NODES * F_INF + (size_t)N_NODES * HID +
                          4u * N_GRAPH * HID + N_GRAPH);
  o += (size_t)nzero;
  // f16 region (every buffer fully rewritten each call)
  _Float16* fb = (_Float16*)(ws + o);
  size_t ho = 0;
  _Float16* xh      = fb + ho; ho += (size_t)N_NODES * HID;
  _Float16* agg1h   = fb + ho; ho += (size_t)N_NODES * HID;
  _Float16* agg2h   = fb + ho; ho += (size_t)N_NODES * HID;
  _Float16* h1rh    = fb + ho; ho += (size_t)N_NODES * HID;
  _Float16* relw1h  = fb + ho; ho += (size_t)HID * HID;
  _Float16* rootw1h = fb + ho; ho += (size_t)HID * HID;
  _Float16* relw2h  = fb + ho; ho += (size_t)HID * HID;
  _Float16* rootw2h = fb + ho; ho += (size_t)HID * HID;
  _Float16* lin1wh  = fb + ho; ho += (size_t)HID * 2 * HID;
  _Float16* lin2wh  = fb + ho; ho += (size_t)16 * HID;      // rows padded 10 -> 16
  _Float16* gsumh   = fb + ho; ho += (size_t)N_GRAPH * 2 * HID;
  _Float16* l1h     = fb + ho; ho += (size_t)N_GRAPH * HID;

  const int NH = N_NODES * HID;  // 6.4M

  zero_kernel<<<(nzero + 255) / 256, 256, 0, stream>>>(zbase, nzero);

  // weight / input f16 conversions (K-padded, branch-free WMMA feeds)
  cvt_f16_kernel<<<16, 256, 0, stream>>>(c1_rel_w,  relw1h,  HID, F_INF, HID, 0, HID * HID);
  cvt_f16_kernel<<<16, 256, 0, stream>>>(c1_root_w, rootw1h, HID, F_INF, HID, 0, HID * HID);
  cvt_f16_kernel<<<16, 256, 0, stream>>>(c2_rel_w,  relw2h,  HID, HID,   HID, 0, HID * HID);
  cvt_f16_kernel<<<16, 256, 0, stream>>>(c2_root_w, rootw2h, HID, HID,   HID, 0, HID * HID);
  cvt_f16_kernel<<<32, 256, 0, stream>>>(lin1_w, lin1wh, HID, 2 * HID, 2 * HID, 0, HID * 2 * HID);
  cvt_f16_kernel<<<4, 256, 0, stream>>>(lin2_w, lin2wh, N_CLS, HID, HID, 0, 16 * HID);
  cvt_f16_kernel<<<(NH + 255) / 256, 256, 0, stream>>>(x, xh, N_NODES, F_INF, HID, 0, NH);

  edge_w_kernel<<<(N_EDGES + 255) / 256, 256, 0, stream>>>(edge_a, bond_w, bond_b, ew);

  // conv1
  scatter1_kernel<<<(N_EDGES * 64) / 256, 256, 0, stream>>>(x, eidx, ew, agg1);
  cvt_f16_kernel<<<(NH + 255) / 256, 256, 0, stream>>>(agg1, agg1h, N_NODES, F_INF, HID, 0, NH);
  dense_conv_kernel<<<(N_NODES / 16) * 4 / 8, 256, 0, stream>>>(agg1h, xh, relw1h, c1_rel_b,
                                                                rootw1h, h1);
  pool_kernel<<<(NH + 255) / 256, 256, 0, stream>>>(h1, batch, pmax1, psum1, cnt, 1);

  // conv2 (relu fused into gather; root term uses relu(h1) via h1rh)
  scatter2_kernel<<<(N_EDGES * 64) / 256, 256, 0, stream>>>(h1, eidx, ew, agg2);
  cvt_f16_kernel<<<(NH + 255) / 256, 256, 0, stream>>>(agg2, agg2h, N_NODES, HID, HID, 0, NH);
  cvt_f16_kernel<<<(NH + 255) / 256, 256, 0, stream>>>(h1, h1rh, N_NODES, HID, HID, 1, NH);
  dense_conv_kernel<<<(N_NODES / 16) * 4 / 8, 256, 0, stream>>>(agg2h, h1rh, relw2h, c2_rel_b,
                                                                rootw2h, h2);
  pool_kernel<<<(NH + 255) / 256, 256, 0, stream>>>(h2, batch, pmax2, psum2, cnt, 0);

  combine_kernel<<<(N_GRAPH * HID + 255) / 256, 256, 0, stream>>>(pmax1, psum1, pmax2, psum2,
                                                                  cnt, gsumh);
  // heads: 512x128 @ 128x64 (+relu) -> f16, then 512x64 @ 64x10 -> f32 out
  linear_wmma_kernel<<<(N_GRAPH / 16) * 4 / 8, 256, 0, stream>>>(
      gsumh, lin1wh, lin1_b, nullptr, l1h, 2 * HID, HID, HID, 4, 1);
  linear_wmma_kernel<<<(N_GRAPH / 16) * 1 / 8, 256, 0, stream>>>(
      l1h, lin2wh, lin2_b, out, nullptr, HID, N_CLS, N_CLS, 1, 0);
}
